// MinkUNet_86947317940509
// MI455X (gfx1250) — matrix operand order repaired and checked by reading, not compiled, not run on hardware
//
#include <hip/hip_runtime.h>
#include <cstdint>
#include <cstddef>

// ---------------------------------------------------------------------------
// Types for CDNA5 WMMA (wave32): A/B = 16 bf16 per lane, C/D = 8 f32 per lane
// ---------------------------------------------------------------------------
typedef __bf16 bf16_t;
typedef __attribute__((ext_vector_type(16))) bf16_t v16bf;
typedef __attribute__((ext_vector_type(8)))  float  v8f;

__device__ __forceinline__ unsigned short f32_to_bf16(float f) {
  unsigned int u = __float_as_uint(f);
  u += 0x7FFFu + ((u >> 16) & 1u);   // round-to-nearest-even
  return (unsigned short)(u >> 16);
}

// ---------------------------------------------------------------------------
// Gathered sparse-conv GEMM with WMMA, 2M x 2N register blocking:
// one wave computes a 32-row x 32-col tile of D.
// fin : bf16 features, padded [Nin+1, Cp] (row Nin is all-zero, Cp % 32 == 0)
// nbr : [Nout, Koff] neighbor indices; value == Nin means "missing" (zero row)
// wpk : packed bf16 weights in B-fragment order:
//       ((k*CoT + ct)*CB + cb)*512 + lane*16 + e
//       lane<16 : N = ct*16+lane,    K(e) = cb*32 + 0 + (e<8 ? e : e+8)   (e = 0..15)
//       lane>=16: N = ct*16+lane-16, K(e) = cb*32 + 8 + (e<8 ? e : e+8)
// out : f32 [Nout, Cout]
// Per (k,cb): 4 A/B fragment load pairs feed 4 independent
// v_wmma_f32_16x16x32_bf16 chains; gathered A rows amortized over 2 N tiles,
// streamed B fragments amortized over 2 M tiles.
// ---------------------------------------------------------------------------
__global__ __launch_bounds__(32) void k_sconv_wmma(
    const unsigned short* __restrict__ fin, int Cp,
    const int* __restrict__ nbr, int Koff, int Nin, int Nout,
    const unsigned short* __restrict__ wpk,
    float* __restrict__ out, int Cout)
{
  const int lane = threadIdx.x;             // 0..31
  const int mt   = blockIdx.x;              // 32-row M super-tile
  const int cp2  = blockIdx.y;              // 32-col N super-tile
  const int CB   = Cp >> 5;
  const int CoT  = Cout >> 4;
  const int ct0  = cp2 * 2;
  const int m0   = mt * 32;
  const int row0 = m0 + (lane & 15);
  const int row1 = row0 + 16;
  const bool v0 = (row0 < Nout);
  const bool v1 = (row1 < Nout);
  // branchless clamped index bases (select pad row Nin when invalid)
  const size_t ib0 = (size_t)(v0 ? row0 : 0) * Koff;
  const size_t ib1 = (size_t)(v1 ? row1 : 0) * Koff;
  // 16-bit A 16x32 fragment: lanes 0-15 hold K {0..7, 16..23},
  //                          lanes 16-31 hold K {8..15, 24..31}
  const int lo = (lane < 16) ? 0 : 8;
  const size_t bkstride = (size_t)CoT * CB * 512;  // per-k B stride (elements)
  const size_t bcstride = (size_t)CB * 512;        // per-ct B stride (elements)

  v8f acc00 = {}, acc01 = {}, acc10 = {}, acc11 = {};
  const unsigned short* bbase =
      wpk + (size_t)ct0 * bcstride + (size_t)lane * 16;

  // software-pipelined neighbor indices (unconditional loads + select)
  int idx0 = nbr[ib0];  idx0 = v0 ? idx0 : Nin;
  int idx1 = nbr[ib1];  idx1 = v1 ? idx1 : Nin;

  for (int k = 0; k < Koff; ++k) {
    const unsigned short* arow0 = fin + (size_t)idx0 * Cp;
    const unsigned short* arow1 = fin + (size_t)idx1 * Cp;
    const unsigned short* bk0   = bbase + (size_t)k * bkstride;
    const unsigned short* bk1   = bk0 + bcstride;

    if (k + 1 < Koff) {
      int t0 = nbr[ib0 + k + 1];  idx0 = v0 ? t0 : Nin;
      int t1 = nbr[ib1 + k + 1];  idx1 = v1 ? t1 : Nin;
      // prefetch next B tiles and next gathered rows (global_prefetch_b8)
      __builtin_prefetch(bk0 + bkstride, 0, 1);
      __builtin_prefetch(bk1 + bkstride, 0, 1);
      __builtin_prefetch(fin + (size_t)idx0 * Cp, 0, 1);
      __builtin_prefetch(fin + (size_t)idx1 * Cp, 0, 1);
    }

    for (int cb = 0; cb < CB; ++cb) {
      union { uint4 q[2]; v16bf v; } A0, A1, B0, B1;
      const int c0 = cb * 32;
      B0.q[0] = *(const uint4*)(bk0 + (size_t)cb * 512);
      B0.q[1] = *(const uint4*)(bk0 + (size_t)cb * 512 + 8);
      B1.q[0] = *(const uint4*)(bk1 + (size_t)cb * 512);
      B1.q[1] = *(const uint4*)(bk1 + (size_t)cb * 512 + 8);
      A0.q[0] = *(const uint4*)(arow0 + c0 + lo);
      A0.q[1] = *(const uint4*)(arow0 + c0 + lo + 16);
      A1.q[0] = *(const uint4*)(arow1 + c0 + lo);
      A1.q[1] = *(const uint4*)(arow1 + c0 + lo + 16);
      acc00 = __builtin_amdgcn_wmma_f32_16x16x32_bf16(
          false, A0.v, false, B0.v, (short)0, acc00, false, false);
      acc10 = __builtin_amdgcn_wmma_f32_16x16x32_bf16(
          false, A1.v, false, B0.v, (short)0, acc10, false, false);
      acc01 = __builtin_amdgcn_wmma_f32_16x16x32_bf16(
          false, A0.v, false, B1.v, (short)0, acc01, false, false);
      acc11 = __builtin_amdgcn_wmma_f32_16x16x32_bf16(
          false, A1.v, false, B1.v, (short)0, acc11, false, false);
    }
  }

  // C/D layout: VGPR r, lanes 0-15 -> M=r, lanes 16-31 -> M=8+r, N=lane%16
  const int ncol0  = ct0 * 16 + (lane & 15);
  const int ncol1  = ncol0 + 16;
  const int mbase0 = m0 + ((lane < 16) ? 0 : 8);
  const int mbase1 = mbase0 + 16;
#pragma unroll
  for (int r = 0; r < 8; ++r) {
    const int mm = mbase0 + r;
    if (mm < Nout) {
      out[(size_t)mm * Cout + ncol0] = acc00[r];
      out[(size_t)mm * Cout + ncol1] = acc01[r];
    }
  }
#pragma unroll
  for (int r = 0; r < 8; ++r) {
    const int mm = mbase1 + r;
    if (mm < Nout) {
      out[(size_t)mm * Cout + ncol0] = acc10[r];
      out[(size_t)mm * Cout + ncol1] = acc11[r];
    }
  }
}

// Pack f32 weights [K, Cin, Cout] -> bf16 B-fragment order (zero-fill c >= Cin)
__global__ void k_pack_w(const float* __restrict__ w, int K, int Cin, int Cout,
                         int Cp, unsigned short* __restrict__ outp)
{
  const int CoT = Cout >> 4, CB = Cp >> 5;
  const long long total = (long long)K * CoT * CB * 512;
  long long i = (long long)blockIdx.x * blockDim.x + threadIdx.x;
  if (i >= total) return;
  const int e    = (int)(i & 15);
  const int lane = (int)((i >> 4) & 31);
  long long t    = i >> 9;
  const int cb = (int)(t % CB); t /= CB;
  const int ct = (int)(t % CoT);
  const int k  = (int)(t / CoT);
  const int co = ct * 16 + (lane & 15);
  const int c  = cb * 32 + ((lane < 16) ? 0 : 8) + ((e < 8) ? e : e + 8);
  float v = (c < Cin) ? w[((long long)k * Cin + c) * Cout + co] : 0.0f;
  outp[i] = f32_to_bf16(v);
}

// ---------------------------------------------------------------------------
// Glue kernels
// ---------------------------------------------------------------------------
__global__ void k_zero16(unsigned short* p, long long n) {
  long long i = (long long)blockIdx.x * blockDim.x + threadIdx.x;
  if (i < n) p[i] = 0;
}

__global__ void k_iota(int* p, int n) {
  int i = blockIdx.x * blockDim.x + threadIdx.x;
  if (i < n) p[i] = i;
}

// write f32 [N,C] into bf16 padded buffer at column offset (buffer pre-zeroed)
__global__ void k_fillbf(const float* __restrict__ x, long long N, int C,
                         unsigned short* __restrict__ y, int Cp, int colOff)
{
  long long i = (long long)blockIdx.x * blockDim.x + threadIdx.x;
  if (i >= N * C) return;
  const int c = (int)(i % C);
  const long long n = i / C;
  y[n * Cp + colOff + c] = f32_to_bf16(x[i]);
}

// per-channel mean/var (population) over N points; one block per channel
__global__ void k_bn_stats(const float* __restrict__ x, int N, int C,
                           float* __restrict__ mv)
{
  const int c = blockIdx.x;
  float s = 0.f, s2 = 0.f;
  for (int n = threadIdx.x; n < N; n += blockDim.x) {
    float v = x[(long long)n * C + c];
    s += v; s2 += v * v;
  }
  __shared__ float sh[256], sh2[256];
  sh[threadIdx.x] = s; sh2[threadIdx.x] = s2;
  __syncthreads();
  for (int o = blockDim.x >> 1; o > 0; o >>= 1) {
    if ((int)threadIdx.x < o) { sh[threadIdx.x] += sh[threadIdx.x + o];
                                sh2[threadIdx.x] += sh2[threadIdx.x + o]; }
    __syncthreads();
  }
  if (threadIdx.x == 0) {
    float m = sh[0] / (float)N;
    mv[c] = m;
    mv[C + c] = sh2[0] / (float)N - m * m;
  }
}

// y = BN(x)*g+b [+resid] [ReLU]; optional f32 out, optional bf16 out at colOff
__global__ void k_bn_apply(const float* __restrict__ x, long long N, int C,
                           const float* __restrict__ mv,
                           const float* __restrict__ g,
                           const float* __restrict__ b,
                           const float* __restrict__ resid, int relu,
                           float* __restrict__ outf,
                           unsigned short* __restrict__ outbf, int Cp, int colOff)
{
  long long i = (long long)blockIdx.x * blockDim.x + threadIdx.x;
  if (i >= N * C) return;
  const int c = (int)(i % C);
  const long long n = i / C;
  const float m = mv[c], v = mv[C + c];
  float y = (x[i] - m) * rsqrtf(v + 1e-5f) * g[c] + b[c];
  if (resid) y += resid[i];
  if (relu)  y = fmaxf(y, 0.f);
  if (outf)  outf[i] = y;
  if (outbf) outbf[n * Cp + colOff + c] = f32_to_bf16(y);
}

// SE: avg over 16 KNN rows
__global__ void k_se_avg(const float* __restrict__ f, const int* __restrict__ knn,
                         long long N, int C, float* __restrict__ avg)
{
  long long i = (long long)blockIdx.x * blockDim.x + threadIdx.x;
  if (i >= N * C) return;
  const int c = (int)(i % C);
  const long long n = i / C;
  float s = 0.f;
#pragma unroll
  for (int k = 0; k < 16; ++k)
    s += f[(long long)knn[n * 16 + k] * C + c];
  avg[i] = s * (1.0f / 16.0f);
}

// SE: s = sigmoid(relu(avg@w1+b1)@w2+b2); out = f*s  (blockDim == C)
__global__ void k_se_apply(const float* __restrict__ f, const float* __restrict__ avg,
                           int C, int H,
                           const float* __restrict__ w1, const float* __restrict__ b1,
                           const float* __restrict__ w2, const float* __restrict__ b2,
                           float* __restrict__ outf,
                           unsigned short* __restrict__ outbf, int Cp)
{
  extern __shared__ float sh[];            // C + H floats
  const long long n = blockIdx.x;
  const int t = threadIdx.x;
  sh[t] = avg[n * C + t];
  __syncthreads();
  float* hbuf = sh + C;
  if (t < H) {
    float a = b1[t];
    for (int c = 0; c < C; ++c) a += sh[c] * w1[c * H + t];
    hbuf[t] = fmaxf(a, 0.f);
  }
  __syncthreads();
  float a = b2[t];
  for (int j = 0; j < H; ++j) a += hbuf[j] * w2[j * C + t];
  const float s = 1.0f / (1.0f + __expf(-a));
  const float y = f[n * C + t] * s;
  if (outf)  outf[n * C + t] = y;
  if (outbf) outbf[n * Cp + t] = f32_to_bf16(y);
}

// classifier: out[n,o] = x[n,:] @ w[:,o] + b[o]
__global__ void k_linear(const float* __restrict__ x, long long N, int Ci,
                         const float* __restrict__ w, const float* __restrict__ b,
                         int Co, float* __restrict__ out)
{
  long long i = (long long)blockIdx.x * blockDim.x + threadIdx.x;
  if (i >= N * Co) return;
  const int o = (int)(i % Co);
  const long long n = i / Co;
  float a = b[o];
  for (int c = 0; c < Ci; ++c) a += x[n * Ci + c] * w[c * Co + o];
  out[i] = a;
}

// ---------------------------------------------------------------------------
// Host orchestration
// ---------------------------------------------------------------------------
namespace {

constexpr int N0 = 20000, N1 = 7000, N2 = 2500, N3 = 900, N4 = 320;
constexpr int NCLS = 19;

struct Ctx {
  char* base; size_t size; size_t lo; size_t hi; hipStream_t stream;
};
static inline size_t alup(size_t n) { return (n + 255) & ~(size_t)255; }
static inline void* palloc(Ctx& c, size_t n) { void* p = c.base + c.lo; c.lo += alup(n); return p; }
static inline void* salloc(Ctx& c, size_t n) { c.hi -= alup(n); return c.base + c.hi; }
static inline void* xalloc(Ctx& c, size_t n, bool pers) { return pers ? palloc(c, n) : salloc(c, n); }
static inline int cpad(int C) { return (C + 31) & ~31; }

struct Feat { float* f; unsigned short* bf; int N, C, Cp; };
struct CbrP { const float *w, *g, *b; };
struct ResP { const float *w1,*g1,*b1,*w2,*g2,*b2,*wd,*gd,*bd; bool hasd; };
struct SeP  { const float *w1,*b1,*w2,*b2; };

static inline unsigned gb(long long total, int bs) {
  return (unsigned)((total + bs - 1) / bs);
}

} // namespace

extern "C" void kernel_launch(void* const* d_in, const int* in_sizes, int n_in,
                              void* d_out, int out_size, void* d_ws, size_t ws_size,
                              hipStream_t stream) {
  (void)in_sizes; (void)n_in; (void)out_size;
  Ctx ctx{(char*)d_ws, ws_size, 0, ws_size, stream};

  // ---- inputs (setup_inputs insertion order) ----
  const float* feats   = (const float*)d_in[0];
  const int*   mmap[5] = {(const int*)d_in[1], (const int*)d_in[2], (const int*)d_in[3],
                          (const int*)d_in[4], (const int*)d_in[5]};
  const int*   dmap[4] = {(const int*)d_in[6], (const int*)d_in[7],
                          (const int*)d_in[8], (const int*)d_in[9]};
  const int*   umap[4] = {(const int*)d_in[10], (const int*)d_in[11],
                          (const int*)d_in[12], (const int*)d_in[13]};
  const int*   knn[5]  = {(const int*)d_in[14], (const int*)d_in[15], (const int*)d_in[16],
                          (const int*)d_in[17], (const int*)d_in[18]};

  int pi = 19;
  auto nf    = [&]() { return (const float*)d_in[pi++]; };
  auto rdCbr = [&]() { CbrP p; p.w = nf(); p.g = nf(); p.b = nf(); return p; };
  auto rdRes = [&](bool hasd) {
    ResP p; p.w1=nf(); p.g1=nf(); p.b1=nf(); p.w2=nf(); p.g2=nf(); p.b2=nf();
    p.hasd = hasd;
    if (hasd) { p.wd=nf(); p.gd=nf(); p.bd=nf(); } else { p.wd=p.gd=p.bd=nullptr; }
    return p;
  };
  auto rdSe = [&]() { SeP p; p.w1=nf(); p.b1=nf(); p.w2=nf(); p.b2=nf(); return p; };

  CbrP stem1 = rdCbr(), stem2 = rdCbr(); SeP sestem = rdSe();
  CbrP down1 = rdCbr(); ResP s1r1 = rdRes(false), s1r2 = rdRes(false); SeP se1 = rdSe();
  CbrP down2 = rdCbr(); ResP s2r1 = rdRes(true),  s2r2 = rdRes(false); SeP se2 = rdSe();
  CbrP down3 = rdCbr(); ResP s3r1 = rdRes(true),  s3r2 = rdRes(false); SeP se3 = rdSe();
  CbrP down4 = rdCbr(); ResP s4r1 = rdRes(true),  s4r2 = rdRes(false); SeP se4 = rdSe();
  CbrP up1 = rdCbr(); ResP u1r1 = rdRes(true), u1r2 = rdRes(false); SeP seu1 = rdSe();
  CbrP up2 = rdCbr(); ResP u2r1 = rdRes(true), u2r2 = rdRes(false); SeP seu2 = rdSe();
  CbrP up3 = rdCbr(); ResP u3r1 = rdRes(true), u3r2 = rdRes(false); SeP seu3 = rdSe();
  CbrP up4 = rdCbr(); ResP u4r1 = rdRes(true), u4r2 = rdRes(false); SeP seu4 = rdSe();
  const float* clsw = nf(); const float* clsb = nf();

  // ---- persistent identity maps (for 1x1 shortcut convs) ----
  const int Ns[5] = {N0, N1, N2, N3, N4};
  int* ids[5];
  for (int l = 0; l < 5; ++l) {
    ids[l] = (int*)palloc(ctx, (size_t)Ns[l] * 4);
    k_iota<<<gb(Ns[l], 256), 256, 0, stream>>>(ids[l], Ns[l]);
  }
  auto idmap = [&](int N) {
    for (int l = 0; l < 5; ++l) if (Ns[l] == N) return ids[l];
    return ids[0];
  };

  // ---- primitive ops ----
  auto mkbf = [&](int N, int C, bool pers) {
    const int Cp = cpad(C);
    const long long n = (long long)(N + 1) * Cp;
    unsigned short* p = (unsigned short*)xalloc(ctx, (size_t)n * 2, pers);
    k_zero16<<<gb(n, 256), 256, 0, stream>>>(p, n);
    return p;
  };
  auto do_pack = [&](const float* w, int K, int Cin, int Cout) {
    const int Cp = cpad(Cin);
    const long long n = (long long)K * Cp * Cout;
    unsigned short* p = (unsigned short*)salloc(ctx, (size_t)n * 2);
    k_pack_w<<<gb(n, 256), 256, 0, stream>>>(w, K, Cin, Cout, Cp, p);
    return p;
  };
  auto do_sconv = [&](const unsigned short* fbf, int Cp, const int* nbr, int K,
                      int Nin, int Nout, const unsigned short* wp,
                      float* out, int Cout) {
    dim3 g((Nout + 31) / 32, Cout / 32);
    k_sconv_wmma<<<g, 32, 0, stream>>>(fbf, Cp, nbr, K, Nin, Nout, wp, out, Cout);
  };
  auto do_bn = [&](const float* x, int N, int C, const float* g, const float* b,
                   const float* resid, bool relu, float* outf,
                   unsigned short* outbf, int Cp, int colOff) {
    float* mv = (float*)salloc(ctx, (size_t)2 * C * 4);
    k_bn_stats<<<C, 256, 0, stream>>>(x, N, C, mv);
    const long long t = (long long)N * C;
    k_bn_apply<<<gb(t, 256), 256, 0, stream>>>(x, N, C, mv, g, b, resid,
                                               relu ? 1 : 0, outf, outbf, Cp, colOff);
  };

  // ---- composite layers ----
  auto cbr = [&](const Feat& in, const int* nbr, int K, int Nout,
                 const CbrP& p, int Cout, bool pers) {
    float* t = (float*)salloc(ctx, (size_t)Nout * Cout * 4);
    do_sconv(in.bf, in.Cp, nbr, K, in.N, Nout, do_pack(p.w, K, in.C, Cout), t, Cout);
    Feat o; o.N = Nout; o.C = Cout; o.Cp = cpad(Cout);
    o.f = (float*)xalloc(ctx, (size_t)Nout * Cout * 4, pers);
    o.bf = mkbf(Nout, Cout, pers);
    do_bn(t, Nout, Cout, p.g, p.b, nullptr, true, o.f, o.bf, o.Cp, 0);
    return o;
  };
  auto res = [&](const Feat& x, const int* nbr, const ResP& p, int Cout, bool pers) {
    const int N = x.N, Cpo = cpad(Cout);
    float* t1 = (float*)salloc(ctx, (size_t)N * Cout * 4);
    do_sconv(x.bf, x.Cp, nbr, 27, N, N, do_pack(p.w1, 27, x.C, Cout), t1, Cout);
    unsigned short* hbf = mkbf(N, Cout, false);
    do_bn(t1, N, Cout, p.g1, p.b1, nullptr, true, nullptr, hbf, Cpo, 0);
    float* t2 = (float*)salloc(ctx, (size_t)N * Cout * 4);
    do_sconv(hbf, Cpo, nbr, 27, N, N, do_pack(p.w2, 27, Cout, Cout), t2, Cout);
    const float* resid;
    if (p.hasd) {
      float* sc = (float*)salloc(ctx, (size_t)N * Cout * 4);
      do_sconv(x.bf, x.Cp, idmap(N), 1, N, N, do_pack(p.wd, 1, x.C, Cout), sc, Cout);
      do_bn(sc, N, Cout, p.gd, p.bd, nullptr, false, sc, nullptr, 0, 0);
      resid = sc;
    } else {
      resid = x.f;
    }
    Feat o; o.N = N; o.C = Cout; o.Cp = Cpo;
    o.f = (float*)xalloc(ctx, (size_t)N * Cout * 4, pers);
    o.bf = mkbf(N, Cout, pers);
    do_bn(t2, N, Cout, p.g2, p.b2, resid, true, o.f, o.bf, o.Cp, 0);
    return o;
  };
  auto se = [&](const Feat& x, const int* kn, const SeP& p, bool pers) {
    const int N = x.N, C = x.C, H = C / 4;
    float* avg = (float*)salloc(ctx, (size_t)N * C * 4);
    const long long t = (long long)N * C;
    k_se_avg<<<gb(t, 256), 256, 0, stream>>>(x.f, kn, N, C, avg);
    Feat o; o.N = N; o.C = C; o.Cp = cpad(C);
    o.f = (float*)xalloc(ctx, (size_t)N * C * 4, pers);
    o.bf = mkbf(N, C, pers);
    k_se_apply<<<N, C, (size_t)(C + H) * 4, stream>>>(x.f, avg, C, H, p.w1, p.b1,
                                                      p.w2, p.b2, o.f, o.bf, o.Cp);
    return o;
  };
  // decoder block: y = se(res(res(concat(cbr(x_coarse,up), skip), m), m), knn)
  auto decoder = [&](const Feat& xc, const Feat& skip, const int* upm, int Nout,
                     const CbrP& up, int Cup, const ResP& r1, const ResP& r2,
                     const SeP& sp, const int* m27, const int* kn, int Cout, bool pers) {
    ctx.hi = ctx.size;  // reset scratch for this stage
    const int Ccat = Cup + skip.C, Cpc = cpad(Ccat);
    unsigned short* ycat = mkbf(Nout, Ccat, false);
    float* t = (float*)salloc(ctx, (size_t)Nout * Cup * 4);
    do_sconv(xc.bf, xc.Cp, upm, 8, xc.N, Nout, do_pack(up.w, 8, xc.C, Cup), t, Cup);
    do_bn(t, Nout, Cup, up.g, up.b, nullptr, true, nullptr, ycat, Cpc, 0);
    const long long tn = (long long)Nout * skip.C;
    k_fillbf<<<gb(tn, 256), 256, 0, stream>>>(skip.f, Nout, skip.C, ycat, Cpc, Cup);
    Feat yc; yc.f = nullptr; yc.bf = ycat; yc.N = Nout; yc.C = Ccat; yc.Cp = Cpc;
    Feat a = res(yc, m27, r1, Cout, false);
    Feat b = res(a,  m27, r2, Cout, false);
    return se(b, kn, sp, pers);
  };

  // ---------------- forward ----------------
  // stem
  ctx.hi = ctx.size;
  Feat fin; fin.f = nullptr; fin.N = N0; fin.C = 4; fin.Cp = 32;
  fin.bf = mkbf(N0, 4, false);
  {
    const long long t = (long long)N0 * 4;
    k_fillbf<<<gb(t, 256), 256, 0, stream>>>(feats, N0, 4, fin.bf, 32, 0);
  }
  Feat st1 = cbr(fin, mmap[0], 27, N0, stem1, 32, false);
  Feat st2 = cbr(st1, mmap[0], 27, N0, stem2, 32, false);
  Feat x0  = se(st2, knn[0], sestem, true);

  // encoder
  ctx.hi = ctx.size;
  Feat e1 = cbr(x0, dmap[0], 8, N1, down1, 32, false);
  Feat x1 = se(res(res(e1, mmap[1], s1r1, 32, false), mmap[1], s1r2, 32, false),
               knn[1], se1, true);

  ctx.hi = ctx.size;
  Feat e2 = cbr(x1, dmap[1], 8, N2, down2, 32, false);
  Feat x2 = se(res(res(e2, mmap[2], s2r1, 64, false), mmap[2], s2r2, 64, false),
               knn[2], se2, true);

  ctx.hi = ctx.size;
  Feat e3 = cbr(x2, dmap[2], 8, N3, down3, 64, false);
  Feat x3 = se(res(res(e3, mmap[3], s3r1, 128, false), mmap[3], s3r2, 128, false),
               knn[3], se3, true);

  ctx.hi = ctx.size;
  Feat e4 = cbr(x3, dmap[3], 8, N4, down4, 128, false);
  Feat x4 = se(res(res(e4, mmap[4], s4r1, 256, false), mmap[4], s4r2, 256, false),
               knn[4], se4, true);

  // decoder
  Feat y3 = decoder(x4, x3, umap[0], N3, up1, 256, u1r1, u1r2, seu1,
                    mmap[3], knn[3], 256, true);
  Feat y2 = decoder(y3, x2, umap[1], N2, up2, 128, u2r1, u2r2, seu2,
                    mmap[2], knn[2], 128, true);
  Feat y1 = decoder(y2, x1, umap[2], N1, up3,  96, u3r1, u3r2, seu3,
                    mmap[1], knn[1],  96, true);
  Feat y0 = decoder(y1, x0, umap[3], N0, up4,  96, u4r1, u4r2, seu4,
                    mmap[0], knn[0],  96, true);

  // classifier
  {
    const long long t = (long long)N0 * NCLS;
    k_linear<<<gb(t, 256), 256, 0, stream>>>(y0.f, N0, 96, clsw, clsb, NCLS,
                                             (float*)d_out);
  }
}
